// MaskedAttention_4853313044615
// MI455X (gfx1250) — matrix operand order repaired and checked by reading, not compiled, not run on hardware
//
#include <hip/hip_runtime.h>
#include <stdint.h>

typedef __attribute__((ext_vector_type(16))) __bf16 v16bf;
typedef __attribute__((ext_vector_type(8)))  __bf16 v8bf;
typedef __attribute__((ext_vector_type(8)))  float  v8f;

#define EMBED 1024
#define SEQ   2048
#define BATCH 2
#define NHEAD 16
#define HDIM  64
#define NTOK  (BATCH * SEQ)   // 4096

// ---------------------------------------------------------------- WMMA helpers

__device__ __forceinline__ v8f wmma_bf16(v16bf a, v16bf b, v8f c) {
  // D = A(16x32) * B(32x16) + C, fp32 accumulate
  return __builtin_amdgcn_wmma_f32_16x16x32_bf16(false, a, false, b, (short)0, c,
                                                 false, false);
}

// A-operand fragment: lane row = lane&15, half = lane>>4.
// element e (0..7)  -> K = half*8 + e ; element 8+e -> K = 16 + half*8 + e
__device__ __forceinline__ v16bf frag_ld_a(const __bf16* p, int hl) {
  v8bf lo = *(const v8bf*)(p + hl * 8);
  v8bf hi = *(const v8bf*)(p + 16 + hl * 8);
  v16bf f;
#pragma unroll
  for (int i = 0; i < 8; ++i) { f[i] = lo[i]; f[i + 8] = hi[i]; }
  return f;
}

// B-operand fragment: lane col = lane&15, half = lane>>4.
// element e (0..7)  -> K = half*8 + e ; element 8+e -> K = 16 + half*8 + e
__device__ __forceinline__ v16bf frag_ld_b(const __bf16* p, int hl) {
  v8bf lo = *(const v8bf*)(p + hl * 8);
  v8bf hi = *(const v8bf*)(p + 16 + hl * 8);
  v16bf f;
#pragma unroll
  for (int i = 0; i < 8; ++i) { f[i] = lo[i]; f[i + 8] = hi[i]; }
  return f;
}

// ------------------------------------------------- CDNA5 async global->LDS DMA
// ASYNCcnt-tracked direct LDS writes (ISA §15.18, opcode 98). INST_OFFSET adds
// to both the LDS and global addresses, so one (lds, vaddr) pair plus literal
// offsets stages a contiguous 64-byte row segment.

__device__ __forceinline__ void async_stage_row64B(uint32_t lds, const __bf16* g) {
  unsigned long long ga = (unsigned long long)(uintptr_t)g;
  asm volatile("global_load_async_to_lds_b128 %0, %1, off"           :: "v"(lds), "v"(ga) : "memory");
  asm volatile("global_load_async_to_lds_b128 %0, %1, off offset:16" :: "v"(lds), "v"(ga) : "memory");
  asm volatile("global_load_async_to_lds_b128 %0, %1, off offset:32" :: "v"(lds), "v"(ga) : "memory");
  asm volatile("global_load_async_to_lds_b128 %0, %1, off offset:48" :: "v"(lds), "v"(ga) : "memory");
}

__device__ __forceinline__ uint32_t lds_off(const void* p) {
  // flat shared address: [63:32] = shared aperture, [31:0] = LDS byte offset
  return (uint32_t)(uintptr_t)p;
}

// ---------------------------------------------------------------- prep kernels

__global__ void k_cvt_bf16(const float* __restrict__ src, __bf16* __restrict__ dst) {
  int i = (blockIdx.x * blockDim.x + threadIdx.x) * 4;
  float4 v = *(const float4*)(src + i);
  dst[i + 0] = (__bf16)v.x;
  dst[i + 1] = (__bf16)v.y;
  dst[i + 2] = (__bf16)v.z;
  dst[i + 3] = (__bf16)v.w;
}

// Wt[n][k] = (bf16) W[k][n]   for the 4 weight matrices (blockIdx.z selects)
__global__ void k_transpose_w(const float* __restrict__ s0, const float* __restrict__ s1,
                              const float* __restrict__ s2, const float* __restrict__ s3,
                              __bf16* __restrict__ d0, __bf16* __restrict__ d1,
                              __bf16* __restrict__ d2, __bf16* __restrict__ d3) {
  const float* S;
  __bf16* D;
  switch (blockIdx.z) {
    case 0:  S = s0; D = d0; break;
    case 1:  S = s1; D = d1; break;
    case 2:  S = s2; D = d2; break;
    default: S = s3; D = d3; break;
  }
  __shared__ float tile[32][33];
  int bx = blockIdx.x * 32, by = blockIdx.y * 32;
  int tx = threadIdx.x, ty = threadIdx.y;
#pragma unroll
  for (int i = 0; i < 4; ++i)
    tile[ty + i * 8][tx] = S[(size_t)(by + ty + i * 8) * EMBED + bx + tx];
  __syncthreads();
#pragma unroll
  for (int i = 0; i < 4; ++i)
    D[(size_t)(bx + ty + i * 8) * EMBED + by + tx] = (__bf16)tile[tx][ty + i * 8];
}

// ---------------------------------------------------------------- GEMM (WMMA)
// C[M=4096][N=1024] = A[M][K=1024] (bf16) x Bt[N][K] (bf16) + bias
// MODE 0: bf16 out row-major [tok][1024]            (Q, K projections)
// MODE 1: bf16 out transposed per head [b][h][d][s] (V projection)
// MODE 2: fp32 out row-major                        (final Wo projection)
// Double-buffered LDS tiles filled by async global->LDS DMA; 16 WMMAs per
// wave per staged tile (BK = 64 = two WMMA k-steps).

#define BMT 128
#define BNT 128
#define BKT 64
#define LDT 72   // padded LDS row stride (elems); 144B = multiple of 16B
#define KITER (EMBED / BKT)

template <int MODE>
__global__ __launch_bounds__(256) void k_gemm(const __bf16* __restrict__ A,
                                              const __bf16* __restrict__ Bt,
                                              const float* __restrict__ bias,
                                              void* __restrict__ outp) {
  __shared__ __bf16 sA[2][BMT * LDT];
  __shared__ __bf16 sB[2][BNT * LDT];

  const int tid = threadIdx.x;
  const int wave = tid >> 5, lane = tid & 31;
  const int r = lane & 15, hl = lane >> 4;
  const int wm = wave >> 1, wn = wave & 1;     // waves: 4 (M) x 2 (N)
  const int blockM = blockIdx.y * BMT, blockN = blockIdx.x * BNT;

  v8f acc[2][4] = {};   // wave tile: 32 (M) x 64 (N)

  // loader mapping: thread -> (row, 32-elem segment); 64B per async group
  const int lrow = tid >> 1, lseg = (tid & 1) * 32;
  const __bf16* gA = A  + (size_t)(blockM + lrow) * EMBED + lseg;
  const __bf16* gB = Bt + (size_t)(blockN + lrow) * EMBED + lseg;
  const uint32_t dA[2] = { lds_off(&sA[0][lrow * LDT + lseg]),
                           lds_off(&sA[1][lrow * LDT + lseg]) };
  const uint32_t dB[2] = { lds_off(&sB[0][lrow * LDT + lseg]),
                           lds_off(&sB[1][lrow * LDT + lseg]) };

  // prologue: stage tile 0 into buffer 0 (8 async b128 per thread)
  async_stage_row64B(dA[0], gA);
  async_stage_row64B(dB[0], gB);

  for (int it = 0; it < KITER; ++it) {
    const int buf = it & 1;
    if (it + 1 < KITER) {
      // stage next tile into the other buffer (safe: last read of that buffer
      // was fenced by the trailing barrier of the previous iteration)
      async_stage_row64B(dA[buf ^ 1], gA + (it + 1) * BKT);
      async_stage_row64B(dB[buf ^ 1], gB + (it + 1) * BKT);
      asm volatile("s_wait_asynccnt 0x8" ::: "memory");  // tile `it` landed
    } else {
      asm volatile("s_wait_asynccnt 0x0" ::: "memory");
    }
    __syncthreads();

    v16bf af[2][2];
#pragma unroll
    for (int mt = 0; mt < 2; ++mt)
#pragma unroll
      for (int ks = 0; ks < 2; ++ks)
        af[mt][ks] = frag_ld_a(&sA[buf][(wm * 32 + mt * 16 + r) * LDT + ks * 32], hl);
#pragma unroll
    for (int nt = 0; nt < 4; ++nt) {
      const __bf16* brow = &sB[buf][(wn * 64 + nt * 16 + r) * LDT];
      v16bf b0 = frag_ld_b(brow, hl);
      v16bf b1 = frag_ld_b(brow + 32, hl);
#pragma unroll
      for (int mt = 0; mt < 2; ++mt) {
        acc[mt][nt] = wmma_bf16(af[mt][0], b0, acc[mt][nt]);
        acc[mt][nt] = wmma_bf16(af[mt][1], b1, acc[mt][nt]);
      }
    }
    __syncthreads();   // all waves done reading `buf` before it is re-staged
  }

  // epilogue: C layout -> row m = j + 8*hl, col n = lane&15 (per 16x16 tile)
#pragma unroll
  for (int mt = 0; mt < 2; ++mt) {
#pragma unroll
    for (int nt = 0; nt < 4; ++nt) {
      const int n = blockN + wn * 64 + nt * 16 + r;
      const float bv = bias[n];
      const int mBase = blockM + wm * 32 + mt * 16 + hl * 8;
#pragma unroll
      for (int j = 0; j < 8; ++j) {
        const int m = mBase + j;
        const float val = acc[mt][nt][j] + bv;
        if (MODE == 0) {
          ((__bf16*)outp)[(size_t)m * EMBED + n] = (__bf16)val;
        } else if (MODE == 1) {
          const int bb = m >> 11, s = m & (SEQ - 1);
          const int hh = n >> 6, d = n & (HDIM - 1);
          ((__bf16*)outp)[((size_t)(bb * NHEAD + hh) * HDIM + d) * SEQ + s] = (__bf16)val;
        } else {
          ((float*)outp)[(size_t)m * EMBED + n] = val;
        }
      }
    }
  }
}

// ---------------------------------------------------------------- attention
// Per block: one (batch, head, 64-query tile). 4 waves x 16 query rows.
// Flash-attention with online softmax over causal key chunks of 64.

__global__ __launch_bounds__(128) void k_attn(const __bf16* __restrict__ qb,
                                              const __bf16* __restrict__ kb,
                                              const __bf16* __restrict__ vT,
                                              __bf16* __restrict__ ao) {
  const int qB = blockIdx.x, h = blockIdx.y, bb = blockIdx.z;
  const int tid = threadIdx.x, wave = tid >> 5, lane = tid & 31;
  const int r = lane & 15, hl = lane >> 4;

  __shared__ __bf16 sP[4 * 16 * 72];          // wave-private 16x64 P tiles
  __bf16* myP = &sP[wave * 16 * 72];

  const size_t tokBase = (size_t)bb * SEQ;
  const int q0 = qB * 64 + wave * 16;

  v16bf qa0, qa1;
  {
    const __bf16* qp = qb + (tokBase + q0 + r) * EMBED + h * HDIM;
    qa0 = frag_ld_a(qp, hl);
    qa1 = frag_ld_a(qp + 32, hl);
  }

  float mrow[8], lrow[8], alpha[8];
  v8f oacc[4] = {};
#pragma unroll
  for (int j = 0; j < 8; ++j) { mrow[j] = -1e30f; lrow[j] = 0.f; }

  for (int kc = 0; kc <= qB; ++kc) {
    const int kbase = kc * 64;

    // ---- S = Q * K^T (16 x 64), 8 WMMAs
    v8f sacc[4] = {};
#pragma unroll
    for (int nt = 0; nt < 4; ++nt) {
      const __bf16* kp = kb + (tokBase + kbase + nt * 16 + r) * EMBED + h * HDIM;
      v16bf kf0 = frag_ld_b(kp, hl);
      v16bf kf1 = frag_ld_b(kp + 32, hl);
      sacc[nt] = wmma_bf16(qa0, kf0, sacc[nt]);
      sacc[nt] = wmma_bf16(qa1, kf1, sacc[nt]);
    }

    // prefetch next key/value chunk into cache (global_prefetch_b8)
    if (kc < qB) {
      const __bf16* nk = kb + (tokBase + kbase + 64 + lane * 2) * EMBED + h * HDIM;
      __builtin_prefetch(nk, 0, 1);
      __builtin_prefetch(nk + EMBED, 0, 1);
      const __bf16* nv = vT + ((size_t)(bb * NHEAD + h) * HDIM + lane * 2) * SEQ + kbase + 64;
      __builtin_prefetch(nv, 0, 1);
      __builtin_prefetch(nv + SEQ, 0, 1);
    }

    // ---- scale, causal mask, online softmax (row = j + 8*hl, col = r)
    const bool edge = (kc == qB);
#pragma unroll
    for (int j = 0; j < 8; ++j) {
      const int qg = q0 + j + 8 * hl;
      float mc = -1e30f;
#pragma unroll
      for (int nt = 0; nt < 4; ++nt) {
        float s = sacc[nt][j] * 0.125f;                      // 1/sqrt(64)
        if (edge && (kbase + nt * 16 + r) > qg) s = -1e30f;  // causal
        sacc[nt][j] = s;
        mc = fmaxf(mc, s);
      }
#pragma unroll
      for (int mk = 1; mk < 16; mk <<= 1) mc = fmaxf(mc, __shfl_xor(mc, mk, 32));
      const float mnew = fmaxf(mrow[j], mc);
      alpha[j] = __expf(mrow[j] - mnew);
      float rs = 0.f;
#pragma unroll
      for (int nt = 0; nt < 4; ++nt) {
        const float pv = __expf(sacc[nt][j] - mnew);
        rs += pv;
        myP[(j + 8 * hl) * 72 + nt * 16 + r] = (__bf16)pv;   // C-layout -> LDS
      }
#pragma unroll
      for (int mk = 1; mk < 16; mk <<= 1) rs += __shfl_xor(rs, mk, 32);
      lrow[j] = lrow[j] * alpha[j] + rs;
      mrow[j] = mnew;
    }

    // DS ops are in-order per wave; fence the compiler only.
    asm volatile("" ::: "memory");

    // ---- O = O*alpha + P * V (16 x 64), 8 WMMAs
    v16bf pf0 = frag_ld_a(&myP[r * 72], hl);
    v16bf pf1 = frag_ld_a(&myP[r * 72 + 32], hl);
#pragma unroll
    for (int nt = 0; nt < 4; ++nt) {
      const __bf16* vp = vT + ((size_t)(bb * NHEAD + h) * HDIM + nt * 16 + r) * SEQ + kbase;
      v16bf vf0 = frag_ld_b(vp, hl);
      v16bf vf1 = frag_ld_b(vp + 32, hl);
#pragma unroll
      for (int j = 0; j < 8; ++j) oacc[nt][j] *= alpha[j];
      oacc[nt] = wmma_bf16(pf0, vf0, oacc[nt]);
      oacc[nt] = wmma_bf16(pf1, vf1, oacc[nt]);
    }
  }

  // ---- normalize and write attn output [tok][h*64 + d] as bf16
#pragma unroll
  for (int nt = 0; nt < 4; ++nt) {
#pragma unroll
    for (int j = 0; j < 8; ++j) {
      const float o = oacc[nt][j] / lrow[j];
      const size_t tok = tokBase + q0 + j + 8 * hl;
      ao[tok * EMBED + h * HDIM + nt * 16 + r] = (__bf16)o;
    }
  }
}

// ---------------------------------------------------------------- launch

extern "C" void kernel_launch(void* const* d_in, const int* in_sizes, int n_in,
                              void* d_out, int out_size, void* d_ws, size_t ws_size,
                              hipStream_t stream) {
  (void)in_sizes; (void)n_in; (void)out_size; (void)ws_size;
  const float* x  = (const float*)d_in[0];
  const float* Wq = (const float*)d_in[1];
  const float* bq = (const float*)d_in[2];
  const float* Wk = (const float*)d_in[3];
  const float* bk = (const float*)d_in[4];
  const float* Wv = (const float*)d_in[5];
  const float* bv = (const float*)d_in[6];
  const float* Wo = (const float*)d_in[7];
  const float* bo = (const float*)d_in[8];

  __bf16* p = (__bf16*)d_ws;
  const size_t TOKE = (size_t)NTOK * EMBED;    // 4M elems
  const size_t WSZ  = (size_t)EMBED * EMBED;   // 1M elems
  __bf16* xb  = p; p += TOKE;                  // x as bf16
  __bf16* WqT = p; p += WSZ;                   // transposed bf16 weights
  __bf16* WkT = p; p += WSZ;
  __bf16* WvT = p; p += WSZ;
  __bf16* WoT = p; p += WSZ;
  __bf16* qbf = p; p += TOKE;                  // Q [tok][e]
  __bf16* kbf = p; p += TOKE;                  // K [tok][e]
  __bf16* vTb = p; p += TOKE;                  // V [b][h][d][s]
  __bf16* aob = p; p += TOKE;                  // attn out [tok][e]

  k_cvt_bf16<<<(int)(TOKE / 1024), 256, 0, stream>>>(x, xb);
  k_transpose_w<<<dim3(32, 32, 4), dim3(32, 8), 0, stream>>>(
      Wq, Wk, Wv, Wo, WqT, WkT, WvT, WoT);

  dim3 gg(EMBED / BNT, NTOK / BMT);            // (8, 32)
  k_gemm<0><<<gg, 256, 0, stream>>>(xb, WqT, bq, qbf);
  k_gemm<0><<<gg, 256, 0, stream>>>(xb, WkT, bk, kbf);
  k_gemm<1><<<gg, 256, 0, stream>>>(xb, WvT, bv, vTb);

  k_attn<<<dim3(SEQ / 64, NHEAD, BATCH), 128, 0, stream>>>(qbf, kbf, vTb, aob);

  k_gemm<2><<<gg, 256, 0, stream>>>(aob, WoT, bo, d_out);
}
